// Qwen2MoeSparseMoeBlock_32375463477423
// MI455X (gfx1250) — matrix operand order repaired and checked by reading, not compiled, not run on hardware
//
#include <hip/hip_runtime.h>
#include <hip/hip_bf16.h>

// ---------------- problem constants ----------------
#define Bq   2
#define Sq   2048
#define Hd   2048
#define Ed   32
#define Kd   4
#define Fd   1408
#define FSd  5632
#define Td   (Bq*Sq)          // 4096 tokens
#define CAPd 1024
#define EPSf 1e-06f

typedef __attribute__((ext_vector_type(16))) __bf16 bf16x16;
typedef __attribute__((ext_vector_type(8)))  __bf16 bf16x8;
typedef __attribute__((ext_vector_type(2)))  __bf16 bf16x2;
typedef __attribute__((ext_vector_type(8)))  float  f32x8;

__device__ __forceinline__ float fast_rcp(float x) { return __builtin_amdgcn_rcpf(x); }
__device__ __forceinline__ float fast_sigmoid(float x) { return fast_rcp(1.f + __expf(-x)); }

// =====================================================================
// 1) prep: RMSNorm -> xn(bf16), router logits/softmax/top4, shared gate
// =====================================================================
__global__ __launch_bounds__(256)
void prep_kernel(const float* __restrict__ inp,
                 const float* __restrict__ w_norm,
                 const float* __restrict__ w_router,   // [H, E]
                 const float* __restrict__ w_sgate,    // [H, 1]
                 __bf16* __restrict__ xn,              // [T, H]
                 int*   __restrict__ sel,              // [T, K]
                 float* __restrict__ wv,               // [T, K]
                 float* __restrict__ sg)               // [T]
{
    const int t   = blockIdx.x;
    const int tid = threadIdx.x;
    const int h0  = tid * 8;
    const float* xrow = inp + (size_t)t * Hd;

    float x[8];
    float ss = 0.f;
#pragma unroll
    for (int i = 0; i < 8; ++i) { x[i] = xrow[h0 + i]; ss += x[i] * x[i]; }

    __shared__ float red[256];
    red[tid] = ss;
    __syncthreads();
    for (int off = 128; off > 0; off >>= 1) {
        if (tid < off) red[tid] += red[tid + off];
        __syncthreads();
    }
    const float r = rsqrtf(red[0] / (float)Hd + EPSf);

    float xf[8];
#pragma unroll
    for (int i = 0; i < 8; ++i) {
        xf[i] = w_norm[h0 + i] * x[i] * r;
        xn[(size_t)t * Hd + h0 + i] = (__bf16)xf[i];
    }

    // router logits + shared-gate dot via LDS atomics
    __shared__ float lg[33];
    if (tid < 33) lg[tid] = 0.f;
    __syncthreads();

    float part[32];
#pragma unroll
    for (int e = 0; e < 32; ++e) part[e] = 0.f;
#pragma unroll
    for (int i = 0; i < 8; ++i) {
        const float xv = xf[i];
        const float* wr = w_router + (size_t)(h0 + i) * Ed;
#pragma unroll
        for (int e = 0; e < 32; ++e) part[e] += xv * wr[e];
    }
#pragma unroll
    for (int e = 0; e < 32; ++e) atomicAdd(&lg[e], part[e]);
    float sp = 0.f;
#pragma unroll
    for (int i = 0; i < 8; ++i) sp += xf[i] * w_sgate[h0 + i];
    atomicAdd(&lg[32], sp);
    __syncthreads();

    if (tid == 0) {
        float probs[32];
        float mx = lg[0];
        for (int e = 1; e < 32; ++e) mx = fmaxf(mx, lg[e]);
        float den = 0.f;
        for (int e = 0; e < 32; ++e) { probs[e] = __expf(lg[e] - mx); den += probs[e]; }
        const float inv = fast_rcp(den);
        for (int e = 0; e < 32; ++e) probs[e] *= inv;
        // top-4, ties broken by lowest index (matches top_k)
        for (int k = 0; k < Kd; ++k) {
            int   bi = 0;
            float bv = -1.f;
            for (int e = 0; e < 32; ++e)
                if (probs[e] > bv) { bv = probs[e]; bi = e; }
            sel[t * Kd + k] = bi;
            wv [t * Kd + k] = bv;
            probs[bi] = -1.f;
        }
        sg[t] = fast_sigmoid(lg[32]);
    }
}

// =====================================================================
// 2) dispatch: stable capacity assignment (matches stable argsort order)
// =====================================================================
__global__ __launch_bounds__(256)
void dispatch_kernel(const int* __restrict__ sel,
                     int* __restrict__ tk_pos,      // [T*K] slot index or -1
                     int* __restrict__ slot_tok,    // [E*CAP]
                     int* __restrict__ counts)      // [E]
{
    const int e   = blockIdx.x;
    const int tid = threadIdx.x;
    const int TK  = Td * Kd;
    const int per = TK / 256;          // 64
    const int base = tid * per;

    int cnt = 0;
    for (int i = 0; i < per; ++i) cnt += (sel[base + i] == e) ? 1 : 0;

    __shared__ int s[256];
    s[tid] = cnt;
    __syncthreads();
    for (int off = 1; off < 256; off <<= 1) {
        int v = (tid >= off) ? s[tid - off] : 0;
        __syncthreads();
        s[tid] += v;
        __syncthreads();
    }
    int pos = s[tid] - cnt;            // exclusive prefix in flat (t,k) order
    const int total = s[255];

    for (int i = 0; i < per; ++i) {
        const int idx = base + i;
        if (sel[idx] == e) {
            if (pos < CAPd) {
                tk_pos[idx]              = e * CAPd + pos;
                slot_tok[e * CAPd + pos] = idx / Kd;
            } else {
                tk_pos[idx] = -1;
            }
            ++pos;
        }
    }
    if (tid == 0) counts[e] = total < CAPd ? total : CAPd;
}

// =====================================================================
// 3) gather: buf[e][c][:] = xn[token] (zero-fill past count)
// =====================================================================
__global__ __launch_bounds__(256)
void gather_kernel(const __bf16* __restrict__ xn,
                   const int* __restrict__ slot_tok,
                   const int* __restrict__ counts,
                   __bf16* __restrict__ buf)
{
    const int slot = blockIdx.x;           // e*CAP + c
    const int e    = slot / CAPd;
    const int c    = slot - e * CAPd;
    const bool valid = c < counts[e];
    const int t    = valid ? slot_tok[slot] : 0;
    const int h0   = threadIdx.x * 8;
    bf16x8 v = {};
    if (valid) v = *(const bf16x8*)(xn + (size_t)t * Hd + h0);
    *(bf16x8*)(buf + (size_t)slot * Hd + h0) = v;
}

// =====================================================================
// WMMA helpers
// =====================================================================
__device__ __forceinline__ bf16x16 make_frag(bf16x8 lo, bf16x8 hi)
{
    bf16x16 f;
#pragma unroll
    for (int i = 0; i < 8; ++i) { f[i] = lo[i]; f[8 + i] = hi[i]; }
    return f;
}

// A fragment (16x32 bf16) straight from row-major global bf16.
// Layout (ISA 7.12.2): lanes 0-15 rows m0..m0+15, K = 8*khalf + 0..7 in
// VGPR0-3 and K = 16 + 8*khalf + 0..7 in VGPR4-7.
__device__ __forceinline__ bf16x16 load_a_frag(const __bf16* A, int row0, int lda,
                                               int k0, int lane)
{
    const int row   = row0 + (lane & 15);
    const int khalf = lane >> 4;
    const bf16x8* p = (const bf16x8*)(A + (size_t)row * lda + k0 + 8 * khalf);
    return make_frag(p[0], p[2]);   // p[2] = +16 elements
}

// B fragment from N-major LDS tile (ldsB[n][k], row stride 40 bf16):
// K = 8*khalf + 0..7 in elements 0-7, K = 16 + 8*khalf + 0..7 in elements 8-15, col = n_base + (lane&15).
__device__ __forceinline__ bf16x16 load_b_frag(const __bf16* lds, int nbase, int lane)
{
    const int n     = nbase + (lane & 15);
    const int khalf = lane >> 4;
    const bf16x8* p = (const bf16x8*)(lds + (size_t)n * 40 + 8 * khalf);
    return make_frag(p[0], p[2]);
}

#define WMMA_BF16(a, b, c) \
    __builtin_amdgcn_wmma_f32_16x16x32_bf16(false, (a), false, (b), (short)0, (c), false, false)

// =====================================================================
// 4) fused gate+up GEMM with SiLU epilogue:
//    Out[m,n] = silu(A@Wg)[m,n] * (A@Wu)[m,n]   (bf16 out)
//    A: [M,Kdim] bf16 row-major; Wg/Wu: [Kdim,N] f32 row-major.
//    Block tile 64x128, 8 waves (2 M x 4 N), wave tile 32x32, K step 32.
//    Per K step per wave: 2 A frags, 4 B frags, 8 WMMAs.
// =====================================================================
__global__ __launch_bounds__(256)
void gemm_gateup(const __bf16* __restrict__ A, size_t strideA,
                 const float* __restrict__ Wg, const float* __restrict__ Wu,
                 size_t strideW,
                 __bf16* __restrict__ Out, size_t strideOut,
                 int Kdim, int N)
{
    const int e = blockIdx.z;
    A   += (size_t)e * strideA;
    Wg  += (size_t)e * strideW;
    Wu  += (size_t)e * strideW;
    Out += (size_t)e * strideOut;

    const int m0   = blockIdx.y * 64;
    const int n0   = blockIdx.x * 128;
    const int tid  = threadIdx.x;
    const int lane = tid & 31;
    const int w    = tid >> 5;
    const int wm   = w & 1;       // 0..1 -> 32-row slice
    const int wn   = w >> 1;      // 0..3 -> 32-col slice

    __shared__ __align__(16) __bf16 ldsG[128 * 40];
    __shared__ __align__(16) __bf16 ldsU[128 * 40];

    f32x8 cg[2][2] = {};
    f32x8 cu[2][2] = {};

    const int sn = tid & 127;     // staging column (0..127)
    const int sk = (tid >> 7) * 16;  // staging k base (0 or 16)

    for (int k0 = 0; k0 < Kdim; k0 += 32) {
        // stage 32x128 weight tiles (f32 -> bf16), pair-packed b32 LDS stores
#pragma unroll
        for (int i = 0; i < 8; ++i) {
            const int k = sk + i * 2;
            const size_t g0 = (size_t)(k0 + k) * N + (n0 + sn);
            const size_t g1 = g0 + N;
            bf16x2 pg; pg[0] = (__bf16)Wg[g0]; pg[1] = (__bf16)Wg[g1];
            *(bf16x2*)&ldsG[sn * 40 + k] = pg;
            bf16x2 pu; pu[0] = (__bf16)Wu[g0]; pu[1] = (__bf16)Wu[g1];
            *(bf16x2*)&ldsU[sn * 40 + k] = pu;
        }
        __syncthreads();

        bf16x16 a0 = load_a_frag(A, m0 + 32 * wm,      Kdim, k0, lane);
        bf16x16 a1 = load_a_frag(A, m0 + 32 * wm + 16, Kdim, k0, lane);
        bf16x16 bg0 = load_b_frag(ldsG, 32 * wn,      lane);
        bf16x16 bg1 = load_b_frag(ldsG, 32 * wn + 16, lane);
        bf16x16 bu0 = load_b_frag(ldsU, 32 * wn,      lane);
        bf16x16 bu1 = load_b_frag(ldsU, 32 * wn + 16, lane);

        cg[0][0] = WMMA_BF16(a0, bg0, cg[0][0]);
        cg[0][1] = WMMA_BF16(a0, bg1, cg[0][1]);
        cg[1][0] = WMMA_BF16(a1, bg0, cg[1][0]);
        cg[1][1] = WMMA_BF16(a1, bg1, cg[1][1]);
        cu[0][0] = WMMA_BF16(a0, bu0, cu[0][0]);
        cu[0][1] = WMMA_BF16(a0, bu1, cu[0][1]);
        cu[1][0] = WMMA_BF16(a1, bu0, cu[1][0]);
        cu[1][1] = WMMA_BF16(a1, bu1, cu[1][1]);
        __syncthreads();
    }

#pragma unroll
    for (int mi = 0; mi < 2; ++mi) {
#pragma unroll
        for (int ni = 0; ni < 2; ++ni) {
#pragma unroll
            for (int r = 0; r < 8; ++r) {
                const int row = m0 + 32 * wm + 16 * mi + r + 8 * (lane >> 4);
                const int col = n0 + 32 * wn + 16 * ni + (lane & 15);
                const float g = cg[mi][ni][r];
                const float u = cu[mi][ni][r];
                Out[(size_t)row * N + col] = (__bf16)(g * u * fast_sigmoid(g));
            }
        }
    }
}

// =====================================================================
// 5) down GEMM: Out = A @ W  (A bf16 [M,Kdim], W f32 [Kdim,N], out bf16)
//    Same 64x128 block tile, 32x32 wave tile, 4 WMMAs per K step.
// =====================================================================
__global__ __launch_bounds__(256)
void gemm_down(const __bf16* __restrict__ A, size_t strideA,
               const float* __restrict__ W, size_t strideW,
               __bf16* __restrict__ Out, size_t strideOut,
               int Kdim, int N)
{
    const int e = blockIdx.z;
    A   += (size_t)e * strideA;
    W   += (size_t)e * strideW;
    Out += (size_t)e * strideOut;

    const int m0   = blockIdx.y * 64;
    const int n0   = blockIdx.x * 128;
    const int tid  = threadIdx.x;
    const int lane = tid & 31;
    const int w    = tid >> 5;
    const int wm   = w & 1;
    const int wn   = w >> 1;

    __shared__ __align__(16) __bf16 ldsB[128 * 40];

    f32x8 c[2][2] = {};

    const int sn = tid & 127;
    const int sk = (tid >> 7) * 16;

    for (int k0 = 0; k0 < Kdim; k0 += 32) {
#pragma unroll
        for (int i = 0; i < 8; ++i) {
            const int k = sk + i * 2;
            const size_t g0 = (size_t)(k0 + k) * N + (n0 + sn);
            bf16x2 p; p[0] = (__bf16)W[g0]; p[1] = (__bf16)W[g0 + N];
            *(bf16x2*)&ldsB[sn * 40 + k] = p;
        }
        __syncthreads();

        bf16x16 a0 = load_a_frag(A, m0 + 32 * wm,      Kdim, k0, lane);
        bf16x16 a1 = load_a_frag(A, m0 + 32 * wm + 16, Kdim, k0, lane);
        bf16x16 b0 = load_b_frag(ldsB, 32 * wn,      lane);
        bf16x16 b1 = load_b_frag(ldsB, 32 * wn + 16, lane);

        c[0][0] = WMMA_BF16(a0, b0, c[0][0]);
        c[0][1] = WMMA_BF16(a0, b1, c[0][1]);
        c[1][0] = WMMA_BF16(a1, b0, c[1][0]);
        c[1][1] = WMMA_BF16(a1, b1, c[1][1]);
        __syncthreads();
    }

#pragma unroll
    for (int mi = 0; mi < 2; ++mi) {
#pragma unroll
        for (int ni = 0; ni < 2; ++ni) {
#pragma unroll
            for (int r = 0; r < 8; ++r) {
                const int row = m0 + 32 * wm + 16 * mi + r + 8 * (lane >> 4);
                const int col = n0 + 32 * wn + 16 * ni + (lane & 15);
                Out[(size_t)row * N + col] = (__bf16)c[mi][ni][r];
            }
        }
    }
}

// =====================================================================
// 6) combine: out = inp + sum_k wv*y[slot] + sigmoid_gate*sh
// =====================================================================
__global__ __launch_bounds__(256)
void combine_kernel(const float* __restrict__ inp,
                    const float* __restrict__ wv,
                    const int*   __restrict__ tk_pos,
                    const __bf16* __restrict__ y,     // [E*CAP, H]
                    const float* __restrict__ sg,
                    const __bf16* __restrict__ sh,    // [T, H]
                    float* __restrict__ out)
{
    const int t  = blockIdx.x;
    const int h0 = threadIdx.x * 8;
    const float gate = sg[t];

    float acc[8];
#pragma unroll
    for (int i = 0; i < 8; ++i) acc[i] = inp[(size_t)t * Hd + h0 + i];

#pragma unroll
    for (int k = 0; k < Kd; ++k) {
        const int slot = tk_pos[t * Kd + k];
        if (slot >= 0) {
            const float wk = wv[t * Kd + k];
            const bf16x8 yv = *(const bf16x8*)(y + (size_t)slot * Hd + h0);
#pragma unroll
            for (int i = 0; i < 8; ++i) acc[i] += wk * (float)yv[i];
        }
    }
    const bf16x8 sv = *(const bf16x8*)(sh + (size_t)t * Hd + h0);
#pragma unroll
    for (int i = 0; i < 8; ++i) acc[i] += gate * (float)sv[i];

#pragma unroll
    for (int i = 0; i < 8; ++i) out[(size_t)t * Hd + h0 + i] = acc[i];
}

// =====================================================================
// launch
// =====================================================================
extern "C" void kernel_launch(void* const* d_in, const int* in_sizes, int n_in,
                              void* d_out, int out_size, void* d_ws, size_t ws_size,
                              hipStream_t stream)
{
    const float* inp       = (const float*)d_in[0];
    const float* w_norm    = (const float*)d_in[1];
    const float* w_router  = (const float*)d_in[2];
    const float* w_gate    = (const float*)d_in[3];
    const float* w_up      = (const float*)d_in[4];
    const float* w_down    = (const float*)d_in[5];
    const float* sw_gate   = (const float*)d_in[6];
    const float* sw_up     = (const float*)d_in[7];
    const float* sw_down   = (const float*)d_in[8];
    const float* w_sgate   = (const float*)d_in[9];
    float* out             = (float*)d_out;

    char* ws = (char*)d_ws;
    auto alloc = [&](size_t bytes) -> void* {
        void* p = (void*)ws;
        ws += (bytes + 255) & ~(size_t)255;
        return p;
    };

    __bf16* xn       = (__bf16*)alloc((size_t)Td * Hd * 2);
    int*    sel      = (int*)   alloc((size_t)Td * Kd * 4);
    float*  wv       = (float*) alloc((size_t)Td * Kd * 4);
    float*  sg       = (float*) alloc((size_t)Td * 4);
    int*    tk_pos   = (int*)   alloc((size_t)Td * Kd * 4);
    int*    counts   = (int*)   alloc((size_t)Ed * 4);
    int*    slot_tok = (int*)   alloc((size_t)Ed * CAPd * 4);
    __bf16* buf      = (__bf16*)alloc((size_t)Ed * CAPd * Hd * 2);
    __bf16* act      = (__bf16*)alloc((size_t)Ed * CAPd * Fd * 2);
    __bf16* yb       = (__bf16*)alloc((size_t)Ed * CAPd * Hd * 2);
    __bf16* sact     = (__bf16*)alloc((size_t)Td * FSd * 2);
    __bf16* shb      = (__bf16*)alloc((size_t)Td * Hd * 2);
    (void)ws_size; (void)in_sizes; (void)n_in; (void)out_size;

    prep_kernel<<<Td, 256, 0, stream>>>(inp, w_norm, w_router, w_sgate,
                                        xn, sel, wv, sg);
    dispatch_kernel<<<Ed, 256, 0, stream>>>(sel, tk_pos, slot_tok, counts);
    gather_kernel<<<Ed * CAPd, 256, 0, stream>>>(xn, slot_tok, counts, buf);

    // expert gate/up: [E] x (CAP x H) @ (H x F)
    gemm_gateup<<<dim3(Fd / 128, CAPd / 64, Ed), 256, 0, stream>>>(
        buf, (size_t)CAPd * Hd,
        w_gate, w_up, (size_t)Hd * Fd,
        act, (size_t)CAPd * Fd,
        Hd, Fd);

    // expert down: [E] x (CAP x F) @ (F x H)
    gemm_down<<<dim3(Hd / 128, CAPd / 64, Ed), 256, 0, stream>>>(
        act, (size_t)CAPd * Fd,
        w_down, (size_t)Fd * Hd,
        yb, (size_t)CAPd * Hd,
        Fd, Hd);

    // shared gate/up: (T x H) @ (H x FS)
    gemm_gateup<<<dim3(FSd / 128, Td / 64, 1), 256, 0, stream>>>(
        xn, 0,
        sw_gate, sw_up, 0,
        sact, 0,
        Hd, FSd);

    // shared down: (T x FS) @ (FS x H)
    gemm_down<<<dim3(Hd / 128, Td / 64, 1), 256, 0, stream>>>(
        sact, 0,
        sw_down, 0,
        shb, 0,
        FSd, Hd);

    combine_kernel<<<Td, 256, 0, stream>>>(inp, wv, tk_pos, yb, sg, shb, out);
}
